// SEGCN_mamba_Layer_63187558859330
// MI455X (gfx1250) — matrix operand order repaired and checked by reading, not compiled, not run on hardware
//
#include <hip/hip_runtime.h>

// ---------------- problem constants (match reference) ----------------
constexpr int  Dm   = 128;          // feature dim
constexpr int  NPGc = 128;          // nodes per group / seq len
constexpr int  Gc   = 32;           // groups
constexpr int  Nn   = Gc * NPGc;    // 4096 nodes
constexpr int  Ec   = 65536;        // edges
constexpr long NDl  = (long)Nn * Dm;

typedef __attribute__((ext_vector_type(16))) _Float16 v16h;
typedef __attribute__((ext_vector_type(8)))  float    v8f;

__device__ __forceinline__ float leakyf(float x){ return x >= 0.f ? x : 0.02f * x; }
__device__ __forceinline__ float reluf (float x){ return x > 0.f  ? x : 0.f; }
__device__ __forceinline__ float siluf (float x){ return x / (1.f + __expf(-x)); }

// gfx1250 async global->LDS copy (ASYNCcnt-tracked; see ISA 08_async_tensor §4).
// lds_off = low 32 bits of the generic shared pointer (wave-relative LDS byte addr).
__device__ __forceinline__ void async_g2l_b128(unsigned lds_off, const float* g){
  asm volatile("global_load_async_to_lds_b128 %0, %1, off" :: "v"(lds_off), "v"(g) : "memory");
}
__device__ __forceinline__ void async_g2l_b32(unsigned lds_off, const float* g){
  asm volatile("global_load_async_to_lds_b32 %0, %1, off" :: "v"(lds_off), "v"(g) : "memory");
}
__device__ __forceinline__ void wait_async0(){
  asm volatile("s_wait_asynccnt 0x0" ::: "memory");
}

// ---------------- trivial elementwise kernels ----------------
__global__ void k_zero(float* p, long n){
  long i = (long)blockIdx.x*blockDim.x + threadIdx.x; if (i < n) p[i] = 0.f;
}
__global__ void k_copy(float* d, const float* s, long n){
  long i = (long)blockIdx.x*blockDim.x + threadIdx.x; if (i < n) d[i] = s[i];
}
__global__ void k_add2(float* o, const float* a, const float* b, long n){
  long i = (long)blockIdx.x*blockDim.x + threadIdx.x; if (i < n) o[i] = a[i] + b[i];
}
__global__ void k_addto(float* d, const float* s, long n){
  long i = (long)blockIdx.x*blockDim.x + threadIdx.x; if (i < n) d[i] += s[i];
}

// ---------------- generic WMMA GEMM: C = act(A*B(+bias)) ----------------
// ACT: 0 none, 1 leaky(0.02), 2 relu.  TRB: B stored [N,K] (use B^T).
// 256 threads = 8 waves; one wave per 16x16 tile. Requires:
//   M%16==0, K%32==0, tilesN=N/16 with tilesN%8==0 OR tilesN==8, and
//   total tiles % 8 == 0 (all call sites satisfy this) -> every block is a full
//   16-row M-strip: the 16xK f32 A-panel is shared by all 8 waves and is staged
//   into LDS with async b128 DMA; fragments are re-read from LDS and cvt to f16.
template<int ACT, int TRB>
__global__ void k_gemm_wmma(const float* __restrict__ A, const float* __restrict__ B,
                            const float* __restrict__ bias, float* __restrict__ C,
                            int M, int N, int K, long sA, long sB, long sC)
{
  __shared__ __align__(16) float Apan[16 * 256];   // up to K=256 (16 KB)
  const int lane = threadIdx.x & 31;
  const int wave = threadIdx.x >> 5;
  const int hf   = lane >> 4;       // 16-lane half
  const int l15  = lane & 15;
  const float* Ab = A + (long)blockIdx.z * sA;
  const float* Bb = B + (long)blockIdx.z * sB;
  float*       Cb = C + (long)blockIdx.z * sC;

  const int  tilesN = N >> 4;
  const long t0 = (long)blockIdx.x * 8;
  const int  tm = (int)(t0 / tilesN);              // uniform across the block
  const int  tn = (int)(t0 % tilesN) + wave;

  // ---- async DMA of the A panel (16 rows x K) into LDS ----
  const float* Abase = Ab + (long)tm * 16 * K;     // row-major, 16B-aligned rows
  const int vecs = (16 * K) >> 2;                  // number of float4 transfers
  for (int v = threadIdx.x; v < vecs; v += 256) {
    async_g2l_b128((unsigned)(uintptr_t)&Apan[v * 4], Abase + (long)v * 4);
  }
  wait_async0();
  __syncthreads();

  const int ncol = (tn << 4) + l15;
  v8f acc = {};
  for (int kc = 0; kc < K; kc += 32) {
    if (kc + 32 < K) {
      if (TRB) __builtin_prefetch(&Bb[(long)ncol * K + kc + 32], 0, 3);
      else     __builtin_prefetch(&Bb[(long)(kc + 32) * N + ncol], 0, 3);
    }
    v16h af, bf;
    // A fragment from LDS: two contiguous 8-float runs per lane
    const float* ap = &Apan[l15 * K + kc];
#pragma unroll
    for (int j = 0; j < 4; ++j) {                  // VGPR j:   K = hf*8 + 2j (+1)
      af[2*j]       = (_Float16)ap[hf * 8 + 2*j];
      af[2*j+1]     = (_Float16)ap[hf * 8 + 2*j + 1];
      af[2*(j+4)]   = (_Float16)ap[16 + hf * 8 + 2*j];     // VGPR 4..7: K = 16+hf*8+2j
      af[2*(j+4)+1] = (_Float16)ap[16 + hf * 8 + 2*j + 1];
    }
    // B fragment: same K order as A (VGPR j<4: K = hf*8+2j, j>=4: K = 16+hf*8+2(j-4))
#pragma unroll
    for (int j = 0; j < 8; ++j) {
      int kb = kc + hf * 8 + 2 * j + (j >= 4 ? 8 : 0);
      if (TRB) {
        bf[2*j]   = (_Float16)Bb[(long)ncol * K + kb];
        bf[2*j+1] = (_Float16)Bb[(long)ncol * K + kb + 1];
      } else {
        bf[2*j]   = (_Float16)Bb[(long)kb       * N + ncol];
        bf[2*j+1] = (_Float16)Bb[(long)(kb + 1) * N + ncol];
      }
    }
    acc = __builtin_amdgcn_wmma_f32_16x16x32_f16(false, af, false, bf,
                                                 (short)0, acc, false, false);
  }
#pragma unroll
  for (int r = 0; r < 8; ++r) {
    int row = (tm << 4) + r + 8 * hf;              // C/D: VGPR r -> M=r (+8 hi half)
    int col = (tn << 4) + l15;
    float v = acc[r];
    if (bias) v += bias[col];
    if (ACT == 1) v = leakyf(v);
    if (ACT == 2) v = reluf(v);
    Cb[(long)row * N + col] = v;
  }
}

template<int ACT, int TRB>
static void gemm(hipStream_t st, const float* A, const float* B, const float* bias, float* C,
                 int M, int N, int K, int batch = 1, long sA = 0, long sB = 0, long sC = 0)
{
  long tiles = (long)(M / 16) * (N / 16);
  dim3 grid((unsigned)((tiles + 7) / 8), 1, (unsigned)batch);
  k_gemm_wmma<ACT, TRB><<<grid, dim3(256), 0, st>>>(A, B, bias, C, M, N, K, sA, sB, sC);
}

// ---------------- edge pipeline ----------------
__global__ void k_edge_feat(const float* __restrict__ x, const int* __restrict__ src,
                            const int* __restrict__ dst, const float* __restrict__ he,
                            float* __restrict__ feat)
{
  int e = blockIdx.x * blockDim.x + threadIdx.x; if (e >= Ec) return;
  int s = src[e], d = dst[e];
  float dx = x[3*s]-x[3*d], dy = x[3*s+1]-x[3*d+1], dz = x[3*s+2]-x[3*d+2];
  float d2 = dx*dx + dy*dy + dz*dz;
  float* f = feat + (long)e * 32;
  f[0] = __expf(-d2);        f[1] = __expf(-d2 * 0.1f);  f[2] = __expf(-d2 * 0.01f);
  f[3] = __expf(-d2 * 1e-3f);f[4] = __expf(-d2 * 1e-4f);
  const float* h = he + (long)e * 27;
  for (int j = 0; j < 27; ++j) f[5 + j] = h[j];
}

__global__ void k_colstats(const float* __restrict__ X, float* __restrict__ stats, long rows){
  int c = threadIdx.x;                        // 128 columns
  float s = 0.f, s2 = 0.f;
  for (long r = blockIdx.x; r < rows; r += gridDim.x) {
    float v = X[r * 128 + c]; s += v; s2 += v * v;
  }
  atomicAdd(&stats[c], s);
  atomicAdd(&stats[128 + c], s2);
}

__global__ void k_bn_apply(const float* __restrict__ X, const float* __restrict__ stats,
                           const float* __restrict__ g, const float* __restrict__ b,
                           float* __restrict__ out, long n, float invRows){
  long i = (long)blockIdx.x*blockDim.x + threadIdx.x; if (i >= n) return;
  int c = (int)(i & 127);
  float m = stats[c] * invRows;
  float v = stats[128 + c] * invRows - m * m;
  out[i] = (X[i] - m) * rsqrtf(v + 1e-5f) * g[c] + b[c];
}

__global__ void k_edge_w(const float* __restrict__ t, const float* __restrict__ stats,
                         const float* __restrict__ g, const float* __restrict__ b,
                         const float* __restrict__ w2, const float* __restrict__ b2,
                         float* __restrict__ w){
  int e = blockIdx.x * blockDim.x + threadIdx.x; if (e >= Ec) return;
  const float invE = 1.f / (float)Ec;
  float acc = 0.f;
  for (int c = 0; c < 128; ++c) {
    float m  = stats[c] * invE;
    float v  = stats[128 + c] * invE - m * m;
    float tn = (t[(long)e * 128 + c] - m) * rsqrtf(v + 1e-5f) * g[c] + b[c];
    acc += tn * w2[c];
  }
  w[e] = reluf(acc + b2[0]);
}

__global__ void k_deg(const float* w, const int* src, float* deg){
  int e = blockIdx.x*blockDim.x + threadIdx.x; if (e >= Ec) return;
  atomicAdd(&deg[src[e]], w[e]);
}
__global__ void k_dinv(const float* deg, float* dinv){
  int n = blockIdx.x*blockDim.x + threadIdx.x; if (n >= Nn) return;
  float d = deg[n];
  dinv[n] = d > 0.f ? rsqrtf(d + 1e-12f) : 0.f;
}
__global__ void k_nw(const float* w, const float* dinv, const int* src, const int* dst, float* nw){
  int e = blockIdx.x*blockDim.x + threadIdx.x; if (e >= Ec) return;
  nw[e] = w[e] * dinv[src[e]] * dinv[dst[e]];
}
// out[dst] += nw * v[src]  (one block per edge, 128 feats; resolves as L2 f32 atomics)
__global__ void k_adj(const float* __restrict__ v, float* __restrict__ out,
                      const float* __restrict__ nw, const int* __restrict__ src,
                      const int* __restrict__ dst){
  int e = blockIdx.x, t = threadIdx.x;
  float c = nw[e];
  atomicAdd(&out[(long)dst[e] * 128 + t], c * v[(long)src[e] * 128 + t]);
}
__global__ void k_horner_init(float* acc, const float* tmp0, const float* bt, long n){
  long i = (long)blockIdx.x*blockDim.x + threadIdx.x; if (i >= n) return;
  acc[i] = (1.f / 64.f) * reluf(bt[6]) * tmp0[i];
}
__global__ void k_horner_step(float* acc, const float* prop, const float* tmpk,
                              const float* bt, int j, float cj, long n){
  long i = (long)blockIdx.x*blockDim.x + threadIdx.x; if (i >= n) return;
  acc[i] = acc[i] - prop[i] + cj * reluf(bt[j]) * tmpk[i];
}

// ---------------- mamba ----------------
__global__ void k_gather(const float* __restrict__ h, const int* __restrict__ perm,
                         float* __restrict__ out){
  int i = blockIdx.x, d = threadIdx.x;
  out[(long)i * Dm + d] = h[(long)perm[i] * Dm + d];
}
__global__ void k_conv_silu(const float* __restrict__ xqz, const float* __restrict__ cw,
                            const float* __restrict__ cb, float* __restrict__ xc){
  int n = blockIdx.x, d = threadIdx.x;
  int l = n & (NPGc - 1), base = n - l;
  float s = cb[d];
#pragma unroll
  for (int k = 0; k < 4; ++k) {
    int lp = l - 3 + k;
    if (lp >= 0) s += xqz[(long)(base + lp) * 256 + d] * cw[d * 4 + k];
  }
  xc[(long)n * 128 + d] = siluf(s);
}
__global__ void k_xproj(const float* __restrict__ xc, const float* __restrict__ W,
                        float* __restrict__ dbl){
  __shared__ float row[128];
  int n = blockIdx.x, t = threadIdx.x;              // blockDim = 64
  row[t] = xc[(long)n * 128 + t];
  row[t + 64] = xc[(long)n * 128 + t + 64];
  __syncthreads();
  if (t < 40) {
    float s = 0.f;
    for (int c = 0; c < 128; ++c) s += row[c] * W[c * 40 + t];
    dbl[(long)n * 40 + t] = s;
  }
}
__global__ void k_dt(const float* __restrict__ dbl, const float* __restrict__ dtw,
                     const float* __restrict__ dtb, float* __restrict__ dt){
  __shared__ float r8[8];
  int n = blockIdx.x, d = threadIdx.x;              // blockDim = 128
  if (d < 8) r8[d] = dbl[(long)n * 40 + d];
  __syncthreads();
  float s = dtb[d];
#pragma unroll
  for (int r = 0; r < 8; ++r) s += r8[r] * dtw[r * 128 + d];
  dt[(long)n * 128 + d] = (s > 20.f) ? s : log1pf(__expf(s));   // softplus
}
// selective scan fused with output gate: gated = (scan_y + xc*Dp) * silu(z)
// B/C state matrices are DMA'd straight into LDS via async b32 copies.
__global__ void k_scan(const float* __restrict__ dtv, const float* __restrict__ dbl,
                       const float* __restrict__ xc, const float* __restrict__ xqz,
                       const float* __restrict__ Alog, const float* __restrict__ Dp,
                       float* __restrict__ gated){
  __shared__ __align__(16) float Bs[NPGc * 16];
  __shared__ __align__(16) float Cs[NPGc * 16];
  int g = blockIdx.x, d = threadIdx.x;
  int base = g * NPGc;
  for (int idx = d; idx < NPGc * 16; idx += 128) {
    int l = idx >> 4, s2 = idx & 15;
    async_g2l_b32((unsigned)(uintptr_t)&Bs[idx], &dbl[(long)(base + l) * 40 + 8 + s2]);
    async_g2l_b32((unsigned)(uintptr_t)&Cs[idx], &dbl[(long)(base + l) * 40 + 24 + s2]);
  }
  wait_async0();
  __syncthreads();
  float Ad[16], st[16];
#pragma unroll
  for (int s = 0; s < 16; ++s) { Ad[s] = -__expf(Alog[d * 16 + s]); st[s] = 0.f; }
  float dc = Dp[d];
  for (int l = 0; l < NPGc; ++l) {
    long n = (long)base + l;
    float dtl = dtv[n * 128 + d];
    float xcl = xc[n * 128 + d];
    float y = 0.f;
#pragma unroll
    for (int s = 0; s < 16; ++s) {
      float dA = __expf(dtl * Ad[s]);
      st[s] = dA * st[s] + dtl * Bs[l * 16 + s] * xcl;
      y += st[s] * Cs[l * 16 + s];
    }
    y += xcl * dc;
    float z = xqz[n * 256 + 128 + d];
    gated[n * 128 + d] = y * siluf(z);
  }
}
// y1[perm[i]] = h[perm[i]] + h_local[perm[i]] + mamba_out[i]
__global__ void k_mamba_combine(const float* __restrict__ h, const float* __restrict__ hloc,
                                const float* __restrict__ mout, const int* __restrict__ perm,
                                float* __restrict__ y1){
  int i = blockIdx.x, d = threadIdx.x;
  long r = (long)perm[i] * 128;
  y1[r + d] = h[r + d] + hloc[r + d] + mout[(long)i * 128 + d];
}

// ---------------- attention softmax (block-diagonal => exact 128-wide) ----------------
__global__ void k_softmax128(float* __restrict__ S){
  __shared__ float red[128];
  long row = blockIdx.x; int t = threadIdx.x;
  float v = S[row * 128 + t];
  red[t] = v; __syncthreads();
  for (int s = 64; s > 0; s >>= 1) { if (t < s) red[t] = fmaxf(red[t], red[t + s]); __syncthreads(); }
  float mx = red[0]; __syncthreads();
  float e = __expf(v - mx);
  red[t] = e; __syncthreads();
  for (int s = 64; s > 0; s >>= 1) { if (t < s) red[t] += red[t + s]; __syncthreads(); }
  S[row * 128 + t] = e / red[0];
}

// ---------------- workspace layout (floats) ----------------
constexpr long o_t     = 0;
constexpr long o_feat  = o_t    + (long)Ec * 128;
constexpr long o_w     = o_feat + (long)Ec * 32;
constexpr long o_nw    = o_w    + Ec;
constexpr long o_deg   = o_nw   + Ec;
constexpr long o_dinv  = o_deg  + Nn;
constexpr long o_stats = o_dinv + Nn;
constexpr long o_tmps  = o_stats + 256;
constexpr long o_prop  = o_tmps + 7 * NDl;
constexpr long o_acc   = o_prop + NDl;          // h_local
constexpr long o_hd    = o_acc  + NDl;
constexpr long o_xqz   = o_hd   + NDl;
constexpr long o_xc    = o_xqz  + (long)Nn * 256;
constexpr long o_dbl   = o_xc   + NDl;
constexpr long o_dt    = o_dbl  + (long)Nn * 40;
constexpr long o_gated = o_dt   + NDl;
constexpr long o_mout  = o_gated + NDl;
constexpr long o_y1    = o_mout + NDl;
// attention / FFN phase aliases over the (dead) edge-t region: 20*NDl of room
constexpr long a_Q  = o_t + 0 * NDl;
constexpr long a_K  = o_t + 1 * NDl;
constexpr long a_V  = o_t + 2 * NDl;
constexpr long a_S  = o_t + 3 * NDl;            // G*128*128 == NDl exactly
constexpr long a_CL = o_t + 4 * NDl;
constexpr long a_CR = o_t + 5 * NDl;
constexpr long a_F1 = o_t + 6 * NDl;            // N*256 = 2*NDl
constexpr long a_FO = o_t + 8 * NDl;
constexpr long a_Y2 = o_t + 9 * NDl;

static inline int gblk(long n, int bs){ return (int)((n + bs - 1) / bs); }

extern "C" void kernel_launch(void* const* d_in, const int* in_sizes, int n_in,
                              void* d_out, int out_size, void* d_ws, size_t ws_size,
                              hipStream_t stream)
{
  (void)in_sizes; (void)n_in; (void)out_size; (void)ws_size;
  float* Wf = (float*)d_ws;

  const float* emlp_w1   = (const float*)d_in[6];
  const float* emlp_b1   = (const float*)d_in[7];
  const float* emlp_bn_g = (const float*)d_in[8];
  const float* emlp_bn_b = (const float*)d_in[9];
  const float* emlp_w2   = (const float*)d_in[10];
  const float* emlp_b2   = (const float*)d_in[11];
  const float* bern_temp = (const float*)d_in[12];
  const float* m_in_w    = (const float*)d_in[13];
  const float* m_conv_w  = (const float*)d_in[14];
  const float* m_conv_b  = (const float*)d_in[15];
  const float* m_xproj_w = (const float*)d_in[16];
  const float* m_dt_w    = (const float*)d_in[17];
  const float* m_dt_b    = (const float*)d_in[18];
  const float* m_A_log   = (const float*)d_in[19];
  const float* m_D       = (const float*)d_in[20];
  const float* m_out_w   = (const float*)d_in[21];
  const float* q_w       = (const float*)d_in[22];
  const float* k_w       = (const float*)d_in[23];
  const float* v_w       = (const float*)d_in[24];
  const float* bn1_g     = (const float*)d_in[25];
  const float* bn1_b     = (const float*)d_in[26];
  const float* bn2_g     = (const float*)d_in[27];
  const float* bn2_b     = (const float*)d_in[28];
  const float* ff_w1     = (const float*)d_in[29];
  const float* ff_b1     = (const float*)d_in[30];
  const float* ff_w2     = (const float*)d_in[31];
  const float* ff_b2     = (const float*)d_in[32];

  const float c6[7] = {1.f, 6.f, 15.f, 20.f, 15.f, 6.f, 1.f};

  // -------- per-side: edge MLP -> Bernstein prop -> Mamba -> bn1 --------
  for (int s = 0; s < 2; ++s) {
    const float* h    = (const float*)d_in[0 + s];
    const float* x    = (const float*)d_in[2 + s];
    const float* he   = (const float*)d_in[4 + s];
    const int*   ei   = (const int*)d_in[33 + s];
    const int*   esrc = ei;
    const int*   edst = ei + Ec;
    const int*   perm = (const int*)d_in[37 + s];
    float* hout = (float*)d_out + (long)s * NDl;

    // edge features + edge MLP (WMMA, leaky, bias)
    k_edge_feat<<<Ec / 256, 256, 0, stream>>>(x, esrc, edst, he, Wf + o_feat);
    gemm<1, 0>(stream, Wf + o_feat, emlp_w1, emlp_b1, Wf + o_t, Ec, 128, 32);
    // batchnorm stats over E rows, then edge weight w = relu(tn . w2 + b2)
    k_zero<<<1, 256, 0, stream>>>(Wf + o_stats, 256);
    k_colstats<<<2048, 128, 0, stream>>>(Wf + o_t, Wf + o_stats, (long)Ec);
    k_edge_w<<<Ec / 256, 256, 0, stream>>>(Wf + o_t, Wf + o_stats, emlp_bn_g, emlp_bn_b,
                                           emlp_w2, emlp_b2, Wf + o_w);
    // degree normalization
    k_zero<<<gblk(Nn, 256), 256, 0, stream>>>(Wf + o_deg, Nn);
    k_deg<<<Ec / 256, 256, 0, stream>>>(Wf + o_w, esrc, Wf + o_deg);
    k_dinv<<<Nn / 256, 256, 0, stream>>>(Wf + o_deg, Wf + o_dinv);
    k_nw<<<Ec / 256, 256, 0, stream>>>(Wf + o_w, Wf + o_dinv, esrc, edst, Wf + o_nw);

    // forward: tmps[k] = (I+A)^k h
    k_copy<<<gblk(NDl, 256), 256, 0, stream>>>(Wf + o_tmps, h, NDl);
    for (int k = 1; k <= 6; ++k) {
      float* prev = Wf + o_tmps + (long)(k - 1) * NDl;
      float* cur  = Wf + o_tmps + (long)k * NDl;
      k_copy<<<gblk(NDl, 256), 256, 0, stream>>>(cur, prev, NDl);
      k_adj<<<Ec, 128, 0, stream>>>(prev, cur, Wf + o_nw, esrc, edst);
    }
    // backward Horner over (I-A): acc <- (I-A)acc + c_j*temp_j*tmps[6-j]
    k_horner_init<<<gblk(NDl, 256), 256, 0, stream>>>(Wf + o_acc, Wf + o_tmps, bern_temp, NDl);
    for (int j = 5; j >= 0; --j) {
      k_zero<<<gblk(NDl, 256), 256, 0, stream>>>(Wf + o_prop, NDl);
      k_adj<<<Ec, 128, 0, stream>>>(Wf + o_acc, Wf + o_prop, Wf + o_nw, esrc, edst);
      k_horner_step<<<gblk(NDl, 256), 256, 0, stream>>>(
          Wf + o_acc, Wf + o_prop, Wf + o_tmps + (long)(6 - j) * NDl,
          bern_temp, j, c6[j] / 64.f, NDl);
    }

    // mamba branch (on permuted h)
    k_gather<<<Nn, 128, 0, stream>>>(h, perm, Wf + o_hd);
    gemm<0, 0>(stream, Wf + o_hd, m_in_w, nullptr, Wf + o_xqz, Nn, 256, 128);
    k_conv_silu<<<Nn, 128, 0, stream>>>(Wf + o_xqz, m_conv_w, m_conv_b, Wf + o_xc);
    k_xproj<<<Nn, 64, 0, stream>>>(Wf + o_xc, m_xproj_w, Wf + o_dbl);
    k_dt<<<Nn, 128, 0, stream>>>(Wf + o_dbl, m_dt_w, m_dt_b, Wf + o_dt);
    k_scan<<<Gc, 128, 0, stream>>>(Wf + o_dt, Wf + o_dbl, Wf + o_xc, Wf + o_xqz,
                                   m_A_log, m_D, Wf + o_gated);
    gemm<0, 0>(stream, Wf + o_gated, m_out_w, nullptr, Wf + o_mout, Nn, 128, 128);
    k_mamba_combine<<<Nn, 128, 0, stream>>>(h, Wf + o_acc, Wf + o_mout, perm, Wf + o_y1);

    // bn1 -> hout (d_out half)
    k_zero<<<1, 256, 0, stream>>>(Wf + o_stats, 256);
    k_colstats<<<512, 128, 0, stream>>>(Wf + o_y1, Wf + o_stats, (long)Nn);
    k_bn_apply<<<gblk(NDl, 256), 256, 0, stream>>>(Wf + o_y1, Wf + o_stats,
                                                   bn1_g, bn1_b, hout, NDl, 1.f / Nn);
  }

  // -------- cross attention (block-diagonal, 32 groups of 128) --------
  float* hl = (float*)d_out;
  float* hr = hl + NDl;
  const long gs = (long)NPGc * NPGc;   // 16384 per-group stride
  for (int pass = 0; pass < 2; ++pass) {
    const float* qh = pass == 0 ? hl : hr;   // queries
    const float* kh = pass == 0 ? hr : hl;   // keys/values
    float* cdst = Wf + (pass == 0 ? a_CL : a_CR);
    gemm<1, 0>(stream, qh, q_w, nullptr, Wf + a_Q, Nn, 128, 128);
    gemm<1, 0>(stream, kh, k_w, nullptr, Wf + a_K, Nn, 128, 128);
    gemm<0, 0>(stream, kh, v_w, nullptr, Wf + a_V, Nn, 128, 128);
    gemm<0, 1>(stream, Wf + a_Q, Wf + a_K, nullptr, Wf + a_S,
               NPGc, NPGc, 128, Gc, gs, gs, gs);           // S = Q K^T per group
    k_softmax128<<<Nn, 128, 0, stream>>>(Wf + a_S);
    gemm<0, 0>(stream, Wf + a_S, Wf + a_V, nullptr, cdst,
               NPGc, NPGc, 128, Gc, gs, gs, gs);           // O = P V per group
  }
  k_addto<<<gblk(NDl, 256), 256, 0, stream>>>(hl, Wf + a_CL, NDl);
  k_addto<<<gblk(NDl, 256), 256, 0, stream>>>(hr, Wf + a_CR, NDl);

  // -------- FFN + bn2 per side (in place on d_out) --------
  for (int s = 0; s < 2; ++s) {
    float* hcur = (s == 0) ? hl : hr;
    gemm<2, 0>(stream, hcur, ff_w1, ff_b1, Wf + a_F1, Nn, 256, 128);
    gemm<0, 0>(stream, Wf + a_F1, ff_w2, ff_b2, Wf + a_FO, Nn, 128, 256);
    k_add2<<<gblk(NDl, 256), 256, 0, stream>>>(Wf + a_Y2, hcur, Wf + a_FO, NDl);
    k_zero<<<1, 256, 0, stream>>>(Wf + o_stats, 256);
    k_colstats<<<512, 128, 0, stream>>>(Wf + a_Y2, Wf + o_stats, (long)Nn);
    k_bn_apply<<<gblk(NDl, 256), 256, 0, stream>>>(Wf + a_Y2, Wf + o_stats,
                                                   bn2_g, bn2_b, hcur, NDl, 1.f / Nn);
  }
}
